// Gemma4AudioRelativePosition_5978594476550
// MI455X (gfx1250) — matrix operand order repaired and checked by reading, not compiled, not run on hardware
//
#include <hip/hip_runtime.h>
#include <math.h>

// MI455X / gfx1250: wave32, WMMA 16x16x32 bf16 with f32 accumulate.
typedef __bf16 v16bf __attribute__((ext_vector_type(16)));
typedef float  v8f   __attribute__((ext_vector_type(8)));

#define SB   72               // bf16 LDS row stride: 144 B = 9*16 -> b128-aligned rows
#define BDS  130              // f32 bd LDS row stride (128 + pad)
#define SMEM_BYTES (6 * 128 * SB * 2 + 128 * BDS * 4)   // 177152 B

static __device__ __forceinline__ unsigned short f32_to_bf16(float x) {
  unsigned int u = __float_as_uint(x);
  u += 0x7FFFu + ((u >> 16) & 1u);       // round-to-nearest-even
  return (unsigned short)(u >> 16);
}
static __device__ __forceinline__ float bf16_hi_f32(unsigned short h) {
  return __uint_as_float(((unsigned int)h) << 16);
}

// Split a float4 into bf16 hi/lo and store as two packed dwords (ds_store_b64).
static __device__ __forceinline__ void split_store4(float4 x,
                                                    unsigned short* hi,
                                                    unsigned short* lo) {
  unsigned short h0 = f32_to_bf16(x.x), h1 = f32_to_bf16(x.y);
  unsigned short h2 = f32_to_bf16(x.z), h3 = f32_to_bf16(x.w);
  unsigned short l0 = f32_to_bf16(x.x - bf16_hi_f32(h0));
  unsigned short l1 = f32_to_bf16(x.y - bf16_hi_f32(h1));
  unsigned short l2 = f32_to_bf16(x.z - bf16_hi_f32(h2));
  unsigned short l3 = f32_to_bf16(x.w - bf16_hi_f32(h3));
  uint2 hp, lp;
  hp.x = (unsigned int)h0 | ((unsigned int)h1 << 16);
  hp.y = (unsigned int)h2 | ((unsigned int)h3 << 16);
  lp.x = (unsigned int)l0 | ((unsigned int)l1 << 16);
  lp.y = (unsigned int)l2 | ((unsigned int)l3 << 16);
  *(uint2*)hi = hp;
  *(uint2*)lo = lp;
}

union FragU { unsigned int u[8]; uint4 q[2]; v16bf v; };

// A matrix 16x32 bf16 (ISA 7.12.2): lane m=lane&15, half=lane>>4.
// VGPRs 0-3 = K (half*8 .. half*8+7), VGPRs 4-7 = K (16+half*8 .. 23+half*8):
// two contiguous 16-byte runs -> two ds_load_b128.
static __device__ __forceinline__ v16bf load_A_frag(const unsigned short* buf,
                                                    int row0, int K0, int lane) {
  const int m = lane & 15, half = lane >> 4;
  const uint4* base = (const uint4*)(buf + (row0 + m) * SB + K0);
  FragU f;
  f.q[0] = base[half];
  f.q[1] = base[2 + half];
  return f.v;
}

// B matrix 32x16 bf16: lane n=lane&15, half=lane>>4; same K order as A:
// K (half*8 .. half*8+7) then K (16+half*8 .. 23+half*8) -> two ds_load_b128.
static __device__ __forceinline__ v16bf load_B_frag(const unsigned short* buf,
                                                    int row0, int K0, int lane) {
  const int n = lane & 15, half = lane >> 4;
  const uint4* base = (const uint4*)(buf + (row0 + n) * SB + K0);
  FragU f;
  f.q[0] = base[half];
  f.q[1] = base[2 + half];
  return f.v;
}

static __device__ __forceinline__ v8f wmma_bf16(v16bf a, v16bf b, v8f c) {
  return __builtin_amdgcn_wmma_f32_16x16x32_bf16(false, a, false, b,
                                                 (short)0, c, false, false);
}

// --- Kernel 1: proj[h][p][d] = (sinusoid row p) . W_pos[h*64+d, :] ----------
extern "C" __global__ __launch_bounds__(256)
void proj_kernel(const float* __restrict__ Wpos, float* __restrict__ proj) {
  __shared__ float emb[512];
  const int p = blockIdx.x;            // 0..127 (position = 127 - p)
  const int t = threadIdx.x;           // 0..255
  const float pos = (float)(127 - p);
  const float log_inc = 9.210340371976184f / 255.0f;   // ln(1e4)/255
  float st = pos * expf(-(float)t * log_inc);
  emb[t]       = sinf(st);
  emb[t + 256] = cosf(st);
  __syncthreads();
#pragma unroll 1
  for (int oo = 0; oo < 2; ++oo) {
    int o = t + (oo << 8);
    const float* w = Wpos + (size_t)o * 512;
    float acc = 0.0f;
#pragma unroll 8
    for (int i = 0; i < 512; ++i) acc += emb[i] * w[i];
    proj[(((o >> 6) * 128) + p) * 64 + (o & 63)] = acc;
  }
}

// --- Kernel 2: out[s,c] = Q.K^T + shifted Q.proj^T --------------------------
extern "C" __global__ __launch_bounds__(256, 1)
void attn_kernel(const float* __restrict__ qg, const float* __restrict__ kg,
                 const float* __restrict__ projg, float* __restrict__ outg) {
  extern __shared__ unsigned char smem[];
  unsigned short* s_qhi = (unsigned short*)smem;
  unsigned short* s_qlo = s_qhi + 128 * SB;
  unsigned short* s_phi = s_qlo + 128 * SB;
  unsigned short* s_plo = s_phi + 128 * SB;
  unsigned short* s_khi = s_plo + 128 * SB;
  unsigned short* s_klo = s_khi + 128 * SB;
  float*          s_bd  = (float*)(s_klo + 128 * SB);

  const int tid  = threadIdx.x;
  const int lane = tid & 31;
  const int wave = tid >> 5;                 // 0..7
  const int bid  = blockIdx.x;               // ((b*8+h)*64+n)
  const int n    = bid & 63;
  const int h    = (bid >> 6) & 7;
  const int b    = bid >> 9;

  const float* qbase = qg + ((size_t)((b * 64 + n) * 128) * 8 + h) * 64; // +s*512+d
  const float* kbase = kg + ((size_t)((b * 64 + n) * 256) * 8 + h) * 64; // +c*512+d
  const float* pbase = projg + (size_t)h * (128 * 64);                   // +p*64+d
  float*       obase = outg + (size_t)bid * (128 * 256);                 // +s*256+c

  // Phase 0: stage Q, proj, K-chunk0 into LDS as bf16 hi/lo (b128 loads,
  // b64 packed stores); prefetch K-chunk1 so it is in flight during compute.
#pragma unroll 1
  for (int it = 0; it < 8; ++it) {
    int idx = tid + (it << 8);               // 0..2047
    int r = idx >> 4;                        // row 0..127
    int d = (idx & 15) << 2;                 // col 0,4,...,60
    float4 xq = *(const float4*)(qbase + (size_t)r * 512 + d);
    split_store4(xq, s_qhi + r * SB + d, s_qlo + r * SB + d);
    float4 xp = *(const float4*)(pbase + (size_t)r * 64 + d);
    split_store4(xp, s_phi + r * SB + d, s_plo + r * SB + d);
    float4 xk = *(const float4*)(kbase + (size_t)r * 512 + d);
    split_store4(xk, s_khi + r * SB + d, s_klo + r * SB + d);
    if ((idx & 7) == 0)                      // one prefetch per 128 B
      __builtin_prefetch(kbase + (size_t)(128 + r) * 512 + d, 0, 0);
  }
  __syncthreads();

  const int S  = wave << 4;                  // this wave's 16-row stripe
  const int m8 = (lane >> 4) << 3;           // C/D layout: m = r + 8*(lane>>4)
  const int nn = lane & 15;

  // Persistent A fragments (Q stripe, both K-chunks, hi and lo).
  v16bf a_hi0 = load_A_frag(s_qhi, S, 0,  lane);
  v16bf a_hi1 = load_A_frag(s_qhi, S, 32, lane);
  v16bf a_lo0 = load_A_frag(s_qlo, S, 0,  lane);
  v16bf a_lo1 = load_A_frag(s_qlo, S, 32, lane);

  // term_bd = Q . proj^T  (own-stripe rows only -> no cross-wave barrier)
#pragma unroll 1
  for (int P = 0; P < 8; ++P) {
    v8f acc = {};
    v16bf bh0 = load_B_frag(s_phi, P << 4, 0, lane);
    v16bf bl0 = load_B_frag(s_plo, P << 4, 0, lane);
    acc = wmma_bf16(a_hi0, bh0, acc);
    acc = wmma_bf16(a_lo0, bh0, acc);
    acc = wmma_bf16(a_hi0, bl0, acc);
    v16bf bh1 = load_B_frag(s_phi, P << 4, 32, lane);
    v16bf bl1 = load_B_frag(s_plo, P << 4, 32, lane);
    acc = wmma_bf16(a_hi1, bh1, acc);
    acc = wmma_bf16(a_lo1, bh1, acc);
    acc = wmma_bf16(a_hi1, bl1, acc);
#pragma unroll
    for (int r = 0; r < 8; ++r)
      s_bd[(S + m8 + r) * BDS + (P << 4) + nn] = acc[r];
  }

  // term_ac over 256 key columns in two 128-column LDS passes.
#pragma unroll 1
  for (int chunk = 0; chunk < 2; ++chunk) {
    if (chunk == 1) {
      __syncthreads();   // everyone done reading K-chunk0
#pragma unroll 1
      for (int it = 0; it < 8; ++it) {
        int idx = tid + (it << 8);
        int r = idx >> 4;
        int d = (idx & 15) << 2;
        float4 xk = *(const float4*)(kbase + (size_t)(128 + r) * 512 + d);
        split_store4(xk, s_khi + r * SB + d, s_klo + r * SB + d);
      }
      __syncthreads();
    }
#pragma unroll 1
    for (int Ct = 0; Ct < 8; ++Ct) {
      v8f acc = {};
      v16bf bh0 = load_B_frag(s_khi, Ct << 4, 0, lane);
      v16bf bl0 = load_B_frag(s_klo, Ct << 4, 0, lane);
      acc = wmma_bf16(a_hi0, bh0, acc);
      acc = wmma_bf16(a_lo0, bh0, acc);
      acc = wmma_bf16(a_hi0, bl0, acc);
      v16bf bh1 = load_B_frag(s_khi, Ct << 4, 32, lane);
      v16bf bl1 = load_B_frag(s_klo, Ct << 4, 32, lane);
      acc = wmma_bf16(a_hi1, bh1, acc);
      acc = wmma_bf16(a_lo1, bh1, acc);
      acc = wmma_bf16(a_hi1, bl1, acc);

      const int c = (chunk << 7) + (Ct << 4) + nn;
#pragma unroll
      for (int r = 0; r < 8; ++r) {
        int s = S + m8 + r;
        float v = acc[r];
        int p = c - s;                       // relative-shift band
        if ((unsigned)p < 128u) v += s_bd[s * BDS + p];
        obase[s * 256 + c] = v;
      }
    }
  }
}

extern "C" void kernel_launch(void* const* d_in, const int* in_sizes, int n_in,
                              void* d_out, int out_size, void* d_ws, size_t ws_size,
                              hipStream_t stream) {
  (void)in_sizes; (void)n_in; (void)out_size; (void)ws_size;
  const float* q    = (const float*)d_in[0];   // (4,64,128,8,64) f32
  const float* k    = (const float*)d_in[1];   // (4,64,256,8,64) f32
  const float* wpos = (const float*)d_in[2];   // (512,512) f32
  float* out  = (float*)d_out;                 // (4,8,64,128,256) f32
  float* proj = (float*)d_ws;                  // 128*512 f32 scratch (256 KiB)

  hipFuncSetAttribute(reinterpret_cast<const void*>(attn_kernel),
                      hipFuncAttributeMaxDynamicSharedMemorySize, SMEM_BYTES);

  proj_kernel<<<128, 256, 0, stream>>>(wpos, proj);
  attn_kernel<<<2048, 256, SMEM_BYTES, stream>>>(q, k, proj, out);
}
